// NeuralAstar_37426345017771
// MI455X (gfx1250) — matrix operand order repaired and checked by reading, not compiled, not run on hardware
//
#include <hip/hip_runtime.h>
#include <hip/hip_bf16.h>

typedef __attribute__((ext_vector_type(16))) _Float16 v16h;
typedef __attribute__((ext_vector_type(8)))  float    v8f;

#define B_    32
#define HW_   1024

// ---------------------------------------------------------------------------
// Pack encoder input: X0[b,p,0]=map, X0[b,p,1]=start+goal  (NHWC fp16)
// ---------------------------------------------------------------------------
__global__ void pack_input_kernel(const float* __restrict__ maps,
                                  const float* __restrict__ start,
                                  const float* __restrict__ goal,
                                  _Float16* __restrict__ X0, int n) {
  int i = blockIdx.x * blockDim.x + threadIdx.x;
  if (i < n) {
    X0[i * 2 + 0] = (_Float16)maps[i];
    X0[i * 2 + 1] = (_Float16)(start[i] + goal[i]);
  }
}

// ---------------------------------------------------------------------------
// fp32 -> fp16 weight conversion WITH transpose:
//   in : w[k*Cout + co]       (HWIO flattened, k = (ky*3+kx)*Cin+ci)
//   out: Wt[co*K + k]         (row per out-channel, contiguous in K)
// This makes the per-lane WMMA B-fragment a contiguous 16-half read.
// ---------------------------------------------------------------------------
__global__ void cvt_wt_kernel(const float* __restrict__ w,
                              _Float16* __restrict__ o, int K, int Cout) {
  int i = blockIdx.x * blockDim.x + threadIdx.x;
  if (i < K * Cout) {
    int k = i / Cout, co = i % Cout;
    o[(size_t)co * K + k] = (_Float16)w[i];
  }
}

// ---------------------------------------------------------------------------
// Conv3x3 as implicit GEMM via WMMA f32_16x16x32_f16, templated per layer.
//   M = 16 pixels (half a map row), N = 16 out channels, K = 9*CIN.
// One wave per 16x16 output tile.
// Fragment layouts used here:
//   A (16x32 f16): lane&15 = M row; halves e -> K=e+16*hi (same order as B)
//   B (32x16 f16): lane&15 = N col; halves e   -> K=e+16*hi   (contiguous!)
//   C/D: lane l, vgpr r -> M = r + 8*(l>=16), N = l&15
// For CIN>=32 each 32-K chunk lies inside a single 3x3 tap, so the A
// fragment is two contiguous 8-half runs -> b128 loads.
// ---------------------------------------------------------------------------
template <int CIN, int COUT, int K>
__global__ void __launch_bounds__(32) conv_wmma_t(
    const _Float16* __restrict__ X,   // [B][1024][CIN]  fp16
    const _Float16* __restrict__ Wt,  // [COUT][K]       fp16 (transposed)
    const float*    __restrict__ bias,
    float*          __restrict__ Y)   // [B][1024][COUT] f32
{
  constexpr int NTN = (COUT + 15) / 16;   // power of two (or 1)
  const int lane   = threadIdx.x;
  const int gid    = blockIdx.x;
  const int ct     = gid % NTN;
  const int pt     = (gid / NTN) & 63;
  const int b      = gid / (NTN * 64);
  const int m      = lane & 15;
  const int hiHalf = lane >> 4;
  const int p      = pt * 16 + m;
  const int py     = p >> 5, px = p & 31;

  const _Float16* Xbase = X + (size_t)b * HW_ * CIN;
  const int co = ct * 16 + (lane & 15);
  const _Float16* Wrow = Wt + (size_t)(co < COUT ? co : 0) * K;

  v8f acc = {};

  if constexpr (CIN >= 32) {
    #pragma unroll
    for (int tap = 0; tap < 9; ++tap) {
      const int yy  = py + tap / 3 - 1;
      const int xx  = px + tap % 3 - 1;
      const bool inb = (yy >= 0) && (yy < 32) && (xx >= 0) && (xx < 32);
      const _Float16* src = Xbase + (size_t)(inb ? (yy * 32 + xx) * CIN : 0);

      for (int cc = 0; cc < CIN; cc += 32) {
        // ---- A fragment: two contiguous 8-half runs ----
        v16h a;
        const int ci0 = cc + hiHalf * 16;
        if (inb) {
          #pragma unroll
          for (int e = 0; e < 8; ++e) a[e]     = src[ci0 + e];
          #pragma unroll
          for (int e = 0; e < 8; ++e) a[8 + e] = src[ci0 + 8 + e];
        } else {
          #pragma unroll
          for (int e = 0; e < 16; ++e) a[e] = (_Float16)0.0f;
        }
        // ---- B fragment: 16 contiguous halves of this channel's K row ----
        v16h bf;
        const int kb = tap * CIN + cc + hiHalf * 16;
        if (co < COUT) {
          #pragma unroll
          for (int e = 0; e < 16; ++e) bf[e] = Wrow[kb + e];
        } else {
          #pragma unroll
          for (int e = 0; e < 16; ++e) bf[e] = (_Float16)0.0f;
        }
        acc = __builtin_amdgcn_wmma_f32_16x16x32_f16(
            false, a, false, bf, (short)0, acc, false, false);
      }
    }
  } else {
    // CIN == 2 (layer 0): K = 18 < 32, single zero-padded chunk, gather path.
    v16h a;
    #pragma unroll
    for (int e = 0; e < 16; ++e) {
      const int kloc = e + hiHalf * 16;
      _Float16 v = (_Float16)0.0f;
      if (kloc < K) {
        const int kidx = kloc >> 1;   // log2(CIN=2)
        const int ci   = kloc & 1;
        const int yy   = py + kidx / 3 - 1;
        const int xx   = px + kidx % 3 - 1;
        if (yy >= 0 && yy < 32 && xx >= 0 && xx < 32)
          v = Xbase[(yy * 32 + xx) * CIN + ci];
      }
      a[e] = v;
    }
    v16h bf;
    #pragma unroll
    for (int e = 0; e < 16; ++e) {
      const int k = e + hiHalf * 16;
      bf[e] = (k < K && co < COUT) ? Wrow[k] : (_Float16)0.0f;
    }
    acc = __builtin_amdgcn_wmma_f32_16x16x32_f16(
        false, a, false, bf, (short)0, acc, false, false);
  }

  if (co < COUT) {
    const float bv = bias[co];
    #pragma unroll
    for (int r = 0; r < 8; ++r) {
      const int mm = r + (hiHalf ? 8 : 0);
      const int pp = pt * 16 + mm;
      Y[((size_t)b * HW_ + pp) * COUT + co] = acc[r] + bv;
    }
  }
}

// ---------------------------------------------------------------------------
// LayerNorm over channel axis + relu/sigmoid. One block per (b,pixel).
// ---------------------------------------------------------------------------
__global__ void __launch_bounds__(256) ln_act_kernel(
    const float* __restrict__ Y, const float* __restrict__ sc,
    const float* __restrict__ bb, _Float16* __restrict__ Xn,
    float* __restrict__ costOut, int Cout, int act)
{
  __shared__ float red[256];
  const int bp = blockIdx.x;           // b*1024 + p
  const int c  = threadIdx.x;
  float x = (c < Cout) ? Y[(size_t)bp * Cout + c] : 0.0f;

  red[c] = x; __syncthreads();
  for (int s = 128; s > 0; s >>= 1) {
    if (c < s) red[c] += red[c + s];
    __syncthreads();
  }
  float mean = red[0] / (float)Cout;
  __syncthreads();
  float d = (c < Cout) ? (x - mean) : 0.0f;
  red[c] = d * d; __syncthreads();
  for (int s = 128; s > 0; s >>= 1) {
    if (c < s) red[c] += red[c + s];
    __syncthreads();
  }
  float var = red[0] / (float)Cout;

  if (c < Cout) {
    float yn = d * rsqrtf(var + 1e-6f) * sc[c] + bb[c];
    float a  = act ? (1.0f / (1.0f + __expf(-yn))) : fmaxf(yn, 0.0f);
    if (Xn)      Xn[(size_t)bp * Cout + c] = (_Float16)a;
    if (costOut && c == 0) costOut[bp] = a;
  }
}

// ---------------------------------------------------------------------------
// Differentiable A* forward + backtrack. One workgroup per batch element,
// 1024 threads = 1 thread per cell = 32 waves. Per-cell state in registers;
// argmax via LDS tree reduction; 256 sequential steps; single-thread
// 256-step backtrack over LDS parents.
// ---------------------------------------------------------------------------
__global__ void __launch_bounds__(1024) astar_kernel(
    const float* __restrict__ cost, const float* __restrict__ obst,
    const float* __restrict__ start, const float* __restrict__ goal,
    float* __restrict__ out)
{
  __shared__ float red_val[1024];
  __shared__ int   red_idx[1024];
  __shared__ int   parents_s[1024];
  __shared__ float path_s[1024];
  __shared__ int   s_goal, s_sel;
  __shared__ float s_g2, s_uns;

  const int b    = blockIdx.x;
  const int tid  = threadIdx.x;
  const int base = b * HW_;
  const int py   = tid >> 5, px = tid & 31;

  const float cost_r = cost[base + tid];
  const float obst_r = obst[base + tid];
  const float goal_r = goal[base + tid];
  float open_r = start[base + tid];
  float g      = 0.0f;
  float hist   = 0.0f;
  int   parent;

  if (goal_r > 0.5f) s_goal = tid;
  if (tid == 0) s_uns = 1.0f;
  __syncthreads();

  parent = s_goal;
  const int gy = s_goal >> 5, gx = s_goal & 31;
  float dy = fabsf((float)(py - gy));
  float dx = fabsf((float)(px - gx));
  const float hc   = fmaxf(dy, dx) + 0.001f * sqrtf(dy * dy + dx * dx) + cost_r;
  const float itau = 1.0f / 5.6568542494923806f;  // 1/sqrt(32)

  for (int t = 0; t < 256; ++t) {
    float fe = __expf(-(0.5f * g + 0.5f * hc) * itau) * open_r;

    red_val[tid] = fe; red_idx[tid] = tid;
    __syncthreads();
    for (int s = 512; s > 0; s >>= 1) {
      if (tid < s) {
        float v2 = red_val[tid + s]; int i2 = red_idx[tid + s];
        if (v2 > red_val[tid] || (v2 == red_val[tid] && i2 < red_idx[tid])) {
          red_val[tid] = v2; red_idx[tid] = i2;
        }
      }
      __syncthreads();
    }
    if (tid == 0) s_sel = (s_uns > 0.5f) ? red_idx[0] : -1;
    __syncthreads();

    const int sc = s_sel;
    if (sc == tid) {
      s_g2 = g + cost_r;                 // broadcast g[sel]+cost[sel]
      if (goal_r > 0.5f) s_uns = 0.0f;   // solved
      hist = 1.0f;                       // hist = clip(hist + sel)
    }
    __syncthreads();

    if (sc >= 0) {
      if (sc == tid) open_r = fmaxf(open_r - s_uns, 0.0f);
      int ady = py - (sc >> 5); if (ady < 0) ady = -ady;
      int adx = px - (sc & 31); if (adx < 0) adx = -adx;
      bool nb = (ady <= 1) && (adx <= 1) && (obst_r > 0.5f);
      if (nb) {
        float g2  = s_g2;
        float idx = (1.0f - open_r) * (1.0f - hist) +
                    open_r * ((g > g2) ? 1.0f : 0.0f);
        if (idx > 0.5f) { g = g2; open_r = 1.0f; parent = sc; }
      }
    }
    __syncthreads();
  }

  // ---- backtrack ----
  parents_s[tid] = parent;
  path_s[tid]    = goal_r;
  __syncthreads();
  if (tid == 0) {
    int loc = s_goal;
    for (int t = 0; t < 256; ++t) {
      path_s[loc] = 1.0f;
      loc = parents_s[loc];
    }
  }
  __syncthreads();

  out[base + tid]            = hist;         // hist output
  out[B_ * HW_ + base + tid] = path_s[tid];  // paths output
}

// ---------------------------------------------------------------------------
// Host-side launcher.
// Inputs (setup_inputs insertion order):
//  0 map_designs [32,32,32,1] f32, 1 start_maps, 2 goal_maps,
//  then per layer i: 3+4i conv_w, 4+4i conv_b, 5+4i ln_s, 6+4i ln_b
// ---------------------------------------------------------------------------
extern "C" void kernel_launch(void* const* d_in, const int* in_sizes, int n_in,
                              void* d_out, int out_size, void* d_ws, size_t ws_size,
                              hipStream_t stream) {
  (void)in_sizes; (void)n_in; (void)out_size; (void)ws_size;

  const float* maps  = (const float*)d_in[0];
  const float* start = (const float*)d_in[1];
  const float* goal  = (const float*)d_in[2];

  // workspace layout
  char* ws = (char*)d_ws;
  float*    Yf    = (float*)ws;                                        // 32 MB
  _Float16* Xa    = (_Float16*)(ws + 33554432);                        // 16 MB
  _Float16* Xb    = (_Float16*)(ws + 33554432 + 16777216);             // 16 MB
  _Float16* Wtb   = (_Float16*)(ws + 33554432 + 2 * 16777216);         // 1 MB
  float*    costf = (float*)  (ws + 33554432 + 2 * 16777216 + 1048576);

  static const int CIN[5]  = {2, 32, 64, 128, 256};
  static const int COUT[5] = {32, 64, 128, 256, 1};

  pack_input_kernel<<<(B_ * HW_ + 255) / 256, 256, 0, stream>>>(
      maps, start, goal, Xa, B_ * HW_);

  _Float16* Xcur = Xa;
  _Float16* Xnxt = Xb;
  for (int i = 0; i < 5; ++i) {
    const float* cw  = (const float*)d_in[3 + i * 4 + 0];
    const float* cb  = (const float*)d_in[3 + i * 4 + 1];
    const float* lns = (const float*)d_in[3 + i * 4 + 2];
    const float* lnb = (const float*)d_in[3 + i * 4 + 3];
    const int Cin = CIN[i], Cout = COUT[i], K = 9 * Cin;

    const int nw = K * Cout;
    cvt_wt_kernel<<<(nw + 255) / 256, 256, 0, stream>>>(cw, Wtb, K, Cout);

    const int ntn = (Cout + 15) / 16;
    const int grid = B_ * 64 * ntn;
    switch (i) {
      case 0: conv_wmma_t<2,   32,  18 ><<<grid, 32, 0, stream>>>(Xcur, Wtb, cb, Yf); break;
      case 1: conv_wmma_t<32,  64,  288><<<grid, 32, 0, stream>>>(Xcur, Wtb, cb, Yf); break;
      case 2: conv_wmma_t<64,  128, 576><<<grid, 32, 0, stream>>>(Xcur, Wtb, cb, Yf); break;
      case 3: conv_wmma_t<128, 256, 1152><<<grid, 32, 0, stream>>>(Xcur, Wtb, cb, Yf); break;
      case 4: conv_wmma_t<256, 1,   2304><<<grid, 32, 0, stream>>>(Xcur, Wtb, cb, Yf); break;
    }

    const int act = (i < 3) ? 0 : 1;
    ln_act_kernel<<<B_ * HW_, 256, 0, stream>>>(
        Yf, lns, lnb, (i < 4) ? Xnxt : (_Float16*)nullptr,
        (i == 4) ? costf : (float*)nullptr, Cout, act);

    _Float16* t = Xcur; Xcur = Xnxt; Xnxt = t;
  }

  astar_kernel<<<B_, 1024, 0, stream>>>(costf, maps, start, goal, (float*)d_out);
}
